// QuantQwenMLP_75505525064735
// MI455X (gfx1250) — compile-verified, matrix operand order checked by reading, not run on hardware
//
#include <hip/hip_runtime.h>
#include <cstdint>
#include <cstddef>

// ---------------------------------------------------------------------------
// QuantQwenMLP for gfx1250 (MI455X): int8 activation x int4 weight fake-quant
// MLP implemented exactly as integer WMMA (v_wmma_i32_16x16x64_iu8) + rank-1
// float rescale. Shapes fixed by setup_inputs(): B=4,S=2048,H=2048,I=5632.
// Global->LDS staging uses CDNA5 async copies (ASYNCcnt) + double buffering.
// ---------------------------------------------------------------------------

typedef int v8i __attribute__((ext_vector_type(8)));

#define M_TOK 8192   // B*S
#define HDIM  2048
#define IDIM  5632

// Low 32 bits of a flat pointer into LDS == LDS byte offset (aperture rule:
// flat->LDS truncates to addr[31:0]).
__device__ inline uint32_t ldsOff(const void* p) {
  return (uint32_t)(uintptr_t)p;
}

// 16B async copy global -> LDS, tracked with ASYNCcnt.
__device__ inline void asyncCopy16(uint32_t lds_addr, const int8_t* gptr) {
  asm volatile("global_load_async_to_lds_b128 %0, %1, off"
               :: "v"(lds_addr), "v"(gptr)
               : "memory");
}

// ---------------------------------------------------------------------------
// Per-row symmetric quantization: scale = max(maxabs/QMAX, 1e-8),
// q = clip(rint(v/scale), -QMAX, QMAX). One workgroup per row.
// ---------------------------------------------------------------------------
template <int QMAX>
__global__ __launch_bounds__(256)
void quant_rows_kernel(const float* __restrict__ src,
                       int8_t* __restrict__ dst,
                       float* __restrict__ scales,
                       int cols) {
  const int row = blockIdx.x;
  const float* r = src + (size_t)row * cols;
  __shared__ float red[256];

  float mx = 0.0f;
  for (int c = threadIdx.x; c < cols; c += 256)
    mx = fmaxf(mx, fabsf(r[c]));
  red[threadIdx.x] = mx;
  __syncthreads();
  for (int s = 128; s > 0; s >>= 1) {
    if (threadIdx.x < s)
      red[threadIdx.x] = fmaxf(red[threadIdx.x], red[threadIdx.x + s]);
    __syncthreads();
  }
  const float scale = fmaxf(red[0] / (float)QMAX, 1e-8f);
  if (threadIdx.x == 0) scales[row] = scale;
  const float inv = 1.0f / scale;

  int8_t* d = dst + (size_t)row * cols;
  for (int c = threadIdx.x; c < cols; c += 256) {
    float q = rintf(r[c] * inv);                 // round-to-nearest-even
    q = fminf(fmaxf(q, (float)(-QMAX)), (float)QMAX);
    d[c] = (int8_t)(int)q;
  }
}

// ---------------------------------------------------------------------------
// Load one 16x64 int8 fragment (A or B role) from LDS, stride 64 bytes/row,
// following the ISA 8-bit WMMA VGPR layout:
//   lane<16  : row r, K-chunks {0-7,16-23,32-39,48-55}
//   lane>=16 : row r, K-chunks {8-15,24-31,40-47,56-63}
// -> four 8-byte LDS loads (ds_load_*_b64) per fragment.
// ---------------------------------------------------------------------------
__device__ inline v8i ldsFrag(const int8_t* base, int rc, int lane) {
  const int half = lane >> 4;
  const int r = lane & 15;
  const int8_t* p = base + (rc + r) * 64 + half * 8;
  const int2 g0 = *(const int2*)(p + 0);
  const int2 g1 = *(const int2*)(p + 16);
  const int2 g2 = *(const int2*)(p + 32);
  const int2 g3 = *(const int2*)(p + 48);
  v8i f;
  f[0] = g0.x; f[1] = g0.y; f[2] = g1.x; f[3] = g1.y;
  f[4] = g2.x; f[5] = g2.y; f[6] = g3.x; f[7] = g3.y;
  return f;
}

// ---------------------------------------------------------------------------
// Tiled int8 WMMA GEMM: C[M,N] = dequant( Aq[M,K] . Bq[N,K]^T ).
// FUSED=true: two weight matrices (gate & up), epilogue = silu(g)*u -> h.
// FUSED=false: single weight matrix, epilogue = dequant -> out.
// Block tile 128x64, K-step 64, 8 waves as 4(M) x 2(N), 32x32 per wave.
// Double-buffered LDS fed by global_load_async_to_lds_b128.
// ---------------------------------------------------------------------------
template <bool FUSED>
__global__ __launch_bounds__(256)
void wmma_gemm_kernel(const int8_t* __restrict__ Aq,  // [M,K] int8
                      const float*  __restrict__ sA,  // [M]
                      const int8_t* __restrict__ B0q, // [N,K] int8
                      const float*  __restrict__ sB0, // [N]
                      const int8_t* __restrict__ B1q, // [N,K] int8 (FUSED)
                      const float*  __restrict__ sB1, // [N]   (FUSED)
                      float* __restrict__ Cout,       // [M,N] fp32
                      int N, int K) {
  constexpr int BM = 128, BN = 64, BK = 64;
  constexpr int ASZ = BM * BK;   // 8 KB per stage
  constexpr int BSZ = BN * BK;   // 4 KB per stage
  constexpr int NLOADS = FUSED ? 4 : 3;  // async b128 loads per thread per tile

  __shared__ __align__(16) int8_t As[2 * ASZ];
  __shared__ __align__(16) int8_t B0s[2 * BSZ];
  __shared__ __align__(16) int8_t B1s[FUSED ? 2 * BSZ : 16];
  __shared__ float sAr[BM];
  __shared__ float sB0r[BN];
  __shared__ float sB1r[FUSED ? BN : 1];

  const int tid = threadIdx.x;
  const int lane = tid & 31;
  const int wid = tid >> 5;       // 0..7
  const int wave_m = wid >> 1;    // 0..3 (32-row slice)
  const int wave_n = wid & 1;     // 0..1 (32-col slice)

  const int m0 = blockIdx.y * BM;
  const int n0 = blockIdx.x * BN;

  if (tid < BM) sAr[tid] = sA[m0 + tid];
  if (tid < BN) {
    sB0r[tid] = sB0[n0 + tid];
    if (FUSED) sB1r[tid] = sB1[n0 + tid];
  }

  v8i accG[2][2], accU[2][2];
  const v8i vzero = {0, 0, 0, 0, 0, 0, 0, 0};
#pragma unroll
  for (int i = 0; i < 2; ++i)
#pragma unroll
    for (int j = 0; j < 2; ++j) { accG[i][j] = vzero; accU[i][j] = vzero; }

  // Per-thread staging addresses (16B chunk per async load).
  const int lrow = tid >> 2;        // 0..63
  const int lcol = (tid & 3) * 16;  // 0/16/32/48 byte offset in K-tile

  const int8_t* gA0 = Aq + (size_t)(m0 + lrow) * K + lcol;
  const int8_t* gA1 = Aq + (size_t)(m0 + lrow + 64) * K + lcol;
  const int8_t* gB0 = B0q + (size_t)(n0 + lrow) * K + lcol;
  const int8_t* gB1 = FUSED ? (B1q + (size_t)(n0 + lrow) * K + lcol) : nullptr;

  const uint32_t lA0 = ldsOff(As) + (uint32_t)(lrow * BK + lcol);
  const uint32_t lA1 = ldsOff(As) + (uint32_t)((lrow + 64) * BK + lcol);
  const uint32_t lB0 = ldsOff(B0s) + (uint32_t)(lrow * BK + lcol);
  const uint32_t lB1 = FUSED ? (ldsOff(B1s) + (uint32_t)(lrow * BK + lcol)) : 0u;

  auto issueTile = [&](int k0, int buf) {
    const uint32_t ao = (uint32_t)(buf * ASZ);
    const uint32_t bo = (uint32_t)(buf * BSZ);
    asyncCopy16(lA0 + ao, gA0 + k0);
    asyncCopy16(lA1 + ao, gA1 + k0);
    asyncCopy16(lB0 + bo, gB0 + k0);
    if (FUSED) asyncCopy16(lB1 + bo, gB1 + k0);
  };

  const int NK = K / BK;
  issueTile(0, 0);

  for (int kt = 0; kt < NK; ++kt) {
    const int cur = kt & 1;
    if (kt + 1 < NK) {
      issueTile((kt + 1) * BK, cur ^ 1);
      // Keep the newest batch in flight; wait for the batch feeding this tile.
      asm volatile("s_wait_asynccnt %0" :: "n"(NLOADS) : "memory");
    } else {
      asm volatile("s_wait_asynccnt 0" ::: "memory");
    }
    __syncthreads();  // tile kt resident in LDS for all waves

    const int8_t* Ab = As + cur * ASZ;
    const int8_t* B0b = B0s + cur * BSZ;
    const int8_t* B1b = FUSED ? (B1s + cur * BSZ) : nullptr;

    v8i a[2], b0[2], b1[2];
#pragma unroll
    for (int i = 0; i < 2; ++i)
      a[i] = ldsFrag(Ab, wave_m * 32 + i * 16, lane);
#pragma unroll
    for (int j = 0; j < 2; ++j)
      b0[j] = ldsFrag(B0b, wave_n * 32 + j * 16, lane);
    if (FUSED) {
#pragma unroll
      for (int j = 0; j < 2; ++j)
        b1[j] = ldsFrag(B1b, wave_n * 32 + j * 16, lane);
    }

#pragma unroll
    for (int i = 0; i < 2; ++i)
#pragma unroll
      for (int j = 0; j < 2; ++j) {
        // (sgn_a, A, sgn_b, B, C, reuse_a, reuse_b) — signed x signed int8
        accG[i][j] = __builtin_amdgcn_wmma_i32_16x16x64_iu8(
            true, a[i], true, b0[j], accG[i][j], false, false);
        if (FUSED)
          accU[i][j] = __builtin_amdgcn_wmma_i32_16x16x64_iu8(
              true, a[i], true, b1[j], accU[i][j], false, false);
      }
    __syncthreads();  // all waves done reading buffer `cur` before overwrite
  }

  // Epilogue. C/D layout: lane<16 -> N=lane, M=e; lane>=16 -> N=lane-16, M=e+8.
  const int half = lane >> 4;
  const int r = lane & 15;
#pragma unroll
  for (int i = 0; i < 2; ++i) {
#pragma unroll
    for (int j = 0; j < 2; ++j) {
      const int colT = wave_n * 32 + j * 16 + r;
      const int col = n0 + colT;
      const float sw0 = sB0r[colT];
      const float sw1 = FUSED ? sB1r[colT] : 0.0f;
#pragma unroll
      for (int e = 0; e < 8; ++e) {
        const int rowT = wave_m * 32 + i * 16 + half * 8 + e;
        const int row = m0 + rowT;
        const float sx = sAr[rowT];
        const float g = (float)accG[i][j][e] * sx * sw0;
        float outv;
        if (FUSED) {
          const float u = (float)accU[i][j][e] * sx * sw1;
          outv = (g / (1.0f + __expf(-g))) * u;   // silu(g) * u
        } else {
          outv = g;
        }
        Cout[(size_t)row * N + col] = outv;
      }
    }
  }
}

// ---------------------------------------------------------------------------
// Host-side launch
// ---------------------------------------------------------------------------
static inline size_t alignUp(size_t x, size_t a) { return (x + a - 1) & ~(a - 1); }

extern "C" void kernel_launch(void* const* d_in, const int* in_sizes, int n_in,
                              void* d_out, int out_size, void* d_ws, size_t ws_size,
                              hipStream_t stream) {
  (void)in_sizes; (void)n_in; (void)out_size; (void)ws_size;

  const float* x  = (const float*)d_in[0];  // [M, H]
  const float* Wg = (const float*)d_in[1];  // [I, H]
  const float* Wu = (const float*)d_in[2];  // [I, H]
  const float* Wd = (const float*)d_in[3];  // [H, I]
  float* out = (float*)d_out;               // [M, H]

  char* ws = (char*)d_ws;
  size_t off = 0;
  auto alloc = [&](size_t bytes) -> char* {
    char* p = ws + off;
    off = alignUp(off + bytes, 256);
    return p;
  };

  int8_t* xq  = (int8_t*)alloc((size_t)M_TOK * HDIM);
  float*  sx  = (float*)alloc((size_t)M_TOK * 4);
  int8_t* Wgq = (int8_t*)alloc((size_t)IDIM * HDIM);
  float*  sWg = (float*)alloc((size_t)IDIM * 4);
  int8_t* Wuq = (int8_t*)alloc((size_t)IDIM * HDIM);
  float*  sWu = (float*)alloc((size_t)IDIM * 4);
  int8_t* Wdq = (int8_t*)alloc((size_t)HDIM * IDIM);
  float*  sWd = (float*)alloc((size_t)HDIM * 4);
  float*  h   = (float*)alloc((size_t)M_TOK * IDIM * 4);
  int8_t* hq  = (int8_t*)alloc((size_t)M_TOK * IDIM);
  float*  sh  = (float*)alloc((size_t)M_TOK * 4);

  // 1) quantize activations (int8) and weights (int4-in-int8)
  quant_rows_kernel<127><<<M_TOK, 256, 0, stream>>>(x, xq, sx, HDIM);
  quant_rows_kernel<7><<<IDIM, 256, 0, stream>>>(Wg, Wgq, sWg, HDIM);
  quant_rows_kernel<7><<<IDIM, 256, 0, stream>>>(Wu, Wuq, sWu, HDIM);
  quant_rows_kernel<7><<<HDIM, 256, 0, stream>>>(Wd, Wdq, sWd, IDIM);

  // 2) fused gate/up int8 WMMA GEMM + SwiGLU -> h [M, I] fp32
  {
    dim3 grid(IDIM / 64, M_TOK / 128);
    wmma_gemm_kernel<true><<<grid, 256, 0, stream>>>(
        xq, sx, Wgq, sWg, Wuq, sWu, h, IDIM, HDIM);
  }

  // 3) per-token quantization of h (int8)
  quant_rows_kernel<127><<<M_TOK, 256, 0, stream>>>(h, hq, sh, IDIM);

  // 4) down projection int8 WMMA GEMM -> out [M, H] fp32
  {
    dim3 grid(HDIM / 64, M_TOK / 128);
    wmma_gemm_kernel<false><<<grid, 256, 0, stream>>>(
        hq, sh, Wdq, sWd, nullptr, nullptr, out, HDIM, IDIM);
  }
}
